// PredictiveCodingLayer_14336600834215
// MI455X (gfx1250) — matrix operand order repaired and checked by reading, not compiled, not run on hardware
//
#include <hip/hip_runtime.h>
#include <hip/hip_bf16.h>

// ---------------- problem constants (match reference) ----------------
#define CC        3
#define HH        224
#define WW        224
#define KHH       5
#define KWW       5
#define SHH       3
#define SWW       3
#define NPH       74
#define NPW       74
#define NC        64                 // N_CAUSES (= K of the GEMM)
#define NB        64                 // batch
#define FEAT      75                 // C*KH*KW (= M of the GEMM)
#define POS_PER_B (NPH * NPW)        // 5476
#define TOTAL_POS (NB * POS_PER_B)   // 350464 (= N of the GEMM), divisible by 16
#define NTILES    (TOTAL_POS / 16)   // 21904
#define WPB       8                  // waves per block (256 threads, wave32)
#define NBLOCKS   (NTILES / WPB)     // 2738 exactly

#define SIGMA2_F  10.0f
#define ALPHA_F   0.1f
#define LAM_F     0.1f
#define ERR_N     (NB * FEAT * POS_PER_B)  // 26,284,800
#define R_N       (NB * NC * POS_PER_B)    // 22,429,696

#define A_FRAG_ELEMS (10 * 32 * 16)        // 5 m-tiles x 2 k-chunks x 32 lanes x 16 bf16

// ---------------- CDNA5 WMMA vector types ----------------
typedef __attribute__((ext_vector_type(16))) __bf16 v16bf;
typedef __attribute__((ext_vector_type(8)))  float  v8f;

// hardware tanh (gfx1250 v_tanh_f32); cheap trans-op fallback otherwise
__device__ __forceinline__ float fast_tanh(float x) {
#if __has_builtin(__builtin_amdgcn_tanhf)
    return __builtin_amdgcn_tanhf(x);
#else
    const float xc = __builtin_fminf(__builtin_fmaxf(x, -16.0f), 16.0f);
#if __has_builtin(__builtin_amdgcn_exp2f)
    const float a = __builtin_amdgcn_exp2f(xc * 2.8853900817779268f); // exp(2x)
#else
    const float a = exp2f(xc * 2.8853900817779268f);
#endif
#if __has_builtin(__builtin_amdgcn_rcpf)
    return (a - 1.0f) * __builtin_amdgcn_rcpf(a + 1.0f);
#else
    return (a - 1.0f) / (a + 1.0f);
#endif
#endif
}

// ---------- kernel 0: seed d_out with lam*mean(U^2) + stage bf16 A fragments ----------
// A = U^T staged in d_ws in exactly the per-lane WMMA fragment order, so the
// main kernel does 10 aligned 32-byte vector loads per lane and ZERO conversions.
__global__ __launch_bounds__(256)
void pc_loss_prep(const float* __restrict__ U, float* __restrict__ out,
                  __bf16* __restrict__ aws) {
    // (a) stage A fragments: element e = ((mt*2+kc)*32 + lane)*16 + i
    for (int e = threadIdx.x; e < A_FRAG_ELEMS; e += 256) {
        const int fi   = e >> 9;           // fragment index: mt*2+kc
        const int rem  = e & 511;
        const int lane = rem >> 4;
        const int i    = rem & 15;
        const int mt   = fi >> 1;
        const int kc   = fi & 1;
        const int col  = lane & 15;
        const int half = lane >> 4;
        const int feat = mt * 16 + col;    // padded M: 80 rows, 75 valid
        const int k    = kc * 32 + half * 16 + i;
        const float v  = (feat < FEAT) ? U[k * FEAT + feat] : 0.0f;
        aws[e] = (__bf16)v;                // native v_cvt bf16
    }
    // (b) lam * mean(U^2) -> overwrite d_out (re-init per call)
    __shared__ float red[WPB];
    float s = 0.0f;
    for (int i = threadIdx.x; i < NC * FEAT; i += 256) {
        const float v = U[i];
        s = fmaf(v, v, s);
    }
    #pragma unroll
    for (int off = 16; off > 0; off >>= 1) s += __shfl_down(s, off, 32);
    const int lane = threadIdx.x & 31, wave = threadIdx.x >> 5;
    if (lane == 0) red[wave] = s;
    __syncthreads();
    if (threadIdx.x == 0) {
        float t = 0.0f;
        #pragma unroll
        for (int w = 0; w < WPB; ++w) t += red[w];
        out[0] = LAM_F * t / (float)(NC * FEAT);
    }
}

// ---------- kernel 1: fused GEMM(bf16 WMMA) + hw tanh + loss reduction ----------
__global__ __launch_bounds__(256)
void pc_loss_main(const float* __restrict__ x,
                  const float* __restrict__ r,
                  const __bf16* __restrict__ aws,
                  float* __restrict__ out) {
    // per-block table: feature -> x element offset (kills div/mod chains);
    // reads of 8 consecutive entries vectorize to ds_load_b128.
    __shared__ int offtab[80];
    if (threadIdx.x < 80) {
        const int f = threadIdx.x;
        offtab[f] = (f < FEAT)
            ? ((f / (KHH * KWW)) * HH * WW + ((f % (KHH * KWW)) / KWW) * WW + (f % KWW))
            : 0;  // padded rows read xb[0] harmlessly; weighted by 0 below
    }
    __syncthreads();

    const int lane = threadIdx.x & 31;
    const int wave = threadIdx.x >> 5;
    const int tile = blockIdx.x * WPB + wave;    // 16-wide N-tile of positions
    const int half = lane >> 4;                  // 0: lanes 0-15, 1: lanes 16-31
    const int col  = lane & 15;                  // column (N) owned by this lane

    // flattened output position owned by this lane's column
    const int gpos = tile * 16 + col;
    const int bb   = gpos / POS_PER_B;
    const int pp   = gpos % POS_PER_B;
    const int prow = pp / NPW;
    const int qcol = pp % NPW;

    // ---- A fragments: pre-staged bf16, 10 x 32B aligned vector loads ----
    const v16bf* __restrict__ Af = (const v16bf*)aws;
    v16bf A[5][2];
    #pragma unroll
    for (int mt = 0; mt < 5; ++mt)
        #pragma unroll
        for (int kc = 0; kc < 2; ++kc)
            A[mt][kc] = Af[(mt * 2 + kc) * 32 + lane];

    // ---- B fragments from r (K=64 as two K=32 chunks) + alpha-prior r^2 ----
    // One per-chunk pointer, then 16 loads with compile-time immediate offsets
    // (max 15*5476*4 B < 2^23, fits the signed 24-bit IOFFSET). Lanes 0-15 read
    // 16 consecutive floats per k-row (coalesced); every r element is touched
    // exactly once across the whole grid.
    float r2 = 0.0f;
    v16bf Bf[2];
    const float* __restrict__ rcol = r + ((size_t)bb * NC) * POS_PER_B
                                       + (size_t)prow * NPW + qcol;
    #pragma unroll
    for (int kc = 0; kc < 2; ++kc) {
        const float* __restrict__ rck =
            rcol + (size_t)(kc * 32 + half * 16) * POS_PER_B;  // hoisted addr math
        #pragma unroll
        for (int i = 0; i < 16; ++i) {
            const float v = rck[i * POS_PER_B];   // immediate-offset global_load_b32
            r2 = fmaf(v, v, r2);
            Bf[kc][i] = (__bf16)v;                // pairs -> v_cvt_pk_bf16_f32
        }
    }

    // ---- 16x16 GEMM tiles over K=64 (2 x v_wmma_f32_16x16x32_bf16 each),
    //      branch-free epilogue: v_tanh_f32, patch gather, masked err^2 ----
    float e2 = 0.0f;
    const float* __restrict__ xb = x + (((size_t)bb * CC) * HH + (size_t)prow * SHH) * WW
                                     + (size_t)qcol * SWW;
    #pragma unroll
    for (int mt = 0; mt < 5; ++mt) {
        v8f D = {};
        D = __builtin_amdgcn_wmma_f32_16x16x32_bf16(
                false, A[mt][0], false, Bf[0], (short)0, D, false, false);
        D = __builtin_amdgcn_wmma_f32_16x16x32_bf16(
                false, A[mt][1], false, Bf[1], (short)0, D, false, false);
        // C/D layout: lane -> column n = lane&15; VGPR v -> row v + 8*half
        #pragma unroll
        for (int v = 0; v < 8; ++v) {
            const int feat  = mt * 16 + half * 8 + v;
            const float w   = (feat < FEAT) ? 1.0f : 0.0f;  // mask padded rows
            const float patch = xb[offtab[feat]];
            const float diff  = patch - fast_tanh(D[v]);
            e2 = fmaf(w * diff, diff, e2);
        }
    }

    // ---- wave reduction + one atomic per wave ----
    float val = e2 * ((1.0f / SIGMA2_F) / (float)ERR_N)
              + r2 * (ALPHA_F / (float)R_N);
    #pragma unroll
    for (int off = 16; off > 0; off >>= 1) val += __shfl_down(val, off, 32);
    if (lane == 0) atomicAdd(out, val);
}

// ---------------- launcher ----------------
extern "C" void kernel_launch(void* const* d_in, const int* in_sizes, int n_in,
                              void* d_out, int out_size, void* d_ws, size_t ws_size,
                              hipStream_t stream) {
    const float* x = (const float*)d_in[0];   // (64, 3, 224, 224)
    const float* U = (const float*)d_in[1];   // (64, 3, 5, 5) -> (64, 75)
    const float* r = (const float*)d_in[2];   // (64, 64, 74, 74)
    float* out   = (float*)d_out;             // scalar loss
    __bf16* aws  = (__bf16*)d_ws;             // 10,240 B staged A fragments

    pc_loss_prep<<<1, 256, 0, stream>>>(U, out, aws);
    pc_loss_main<<<NBLOCKS, 256, 0, stream>>>(x, r, aws, out);
}
